// MultiHeadAttention_44856638439436
// MI455X (gfx1250) — matrix operand order repaired and checked by reading, not compiled, not run on hardware
//
#include <hip/hip_runtime.h>

typedef __attribute__((ext_vector_type(16))) __bf16 v16bf;
typedef __attribute__((ext_vector_type(8)))  float  v8f;
typedef __attribute__((ext_vector_type(4)))  unsigned int u32x4;

union Frag16 { v16bf v; u32x4 q[2]; unsigned short s[16]; };

#define S_LEN   2048
#define D_MODEL 1024
#define NHEADS  16
#define DEPTH   64

static __device__ inline unsigned short f2bf_u(float f) {
    union { float f; unsigned u; } x; x.f = f;
    unsigned u = x.u;
    u += 0x7fffu + ((u >> 16) & 1u);   // round-to-nearest-even
    return (unsigned short)(u >> 16);
}

__global__ void cvt_f32_bf16(const float* __restrict__ in,
                             unsigned short* __restrict__ out, int n) {
    int i = blockIdx.x * blockDim.x + threadIdx.x;
    if (i < n) out[i] = f2bf_u(in[i]);
}

// convert + transpose a 1024x1024 weight: out[col][row] = bf16(in[row][col])
__global__ void cvt_f32_bf16_T(const float* __restrict__ in,
                               unsigned short* __restrict__ out) {
    int i = blockIdx.x * blockDim.x + threadIdx.x;
    int row = i >> 10, col = i & 1023;
    out[(size_t)col * 1024 + row] = f2bf_u(in[i]);
}

// C = A(8192x1024 bf16, row-major) x W (given as Bt = W^T, [N][K] bf16) + bias.
// LDS-free; all 8 B fragments batched into registers per K-step so loads overlap WMMA.
// mode 0: store bf16 head-split [B,H,S,64]; mode 1: bf16 [B,H,64,S]; mode 2: fp32 [8192,1024]
__global__ __launch_bounds__(256) void wmma_gemm(
    const unsigned short* __restrict__ A, const unsigned short* __restrict__ Bt,
    const float* __restrict__ bias, unsigned short* __restrict__ outb,
    float* __restrict__ outf, int mode)
{
    const int tid  = threadIdx.x;
    const int wave = tid >> 5;
    const int lane = tid & 31;
    const int lm   = lane & 15;
    const int half = lane >> 4;
    const int rbase = blockIdx.y * 128;
    const int nbase = blockIdx.x * 128;

    const int row  = rbase + wave * 16 + lm;   // A-fragment row owned by this lane
    const int koff = half * 8;                 // A layout: lanes>=16 start at K=8
    const int k2   = half * 8;                 // B layout: lanes>=16 start at K=8
    const unsigned short* arow = A  + (size_t)row * 1024 + koff;
    const unsigned short* bp0  = Bt + (size_t)(nbase + lm) * 1024 + k2;

    v8f z = {0.f,0.f,0.f,0.f,0.f,0.f,0.f,0.f};
    v8f acc[8];
#pragma unroll
    for (int n = 0; n < 8; ++n) acc[n] = z;

    for (int kb = 0; kb < 1024; kb += 32) {
        Frag16 fa;
        fa.q[0] = *(const u32x4*)(arow + kb);        // K koff..koff+7
        fa.q[1] = *(const u32x4*)(arow + kb + 16);   // K koff+16..koff+23
        Frag16 fb[8];
#pragma unroll
        for (int n = 0; n < 8; ++n) {
            const unsigned short* p = bp0 + kb + n * 16384;  // n*16 cols * 1024 K
            fb[n].q[0] = *(const u32x4*)p;
            fb[n].q[1] = *(const u32x4*)(p + 16);
        }
#pragma unroll
        for (int n = 0; n < 8; ++n)
            acc[n] = __builtin_amdgcn_wmma_f32_16x16x32_bf16(
                false, fa.v, false, fb[n].v, (short)0, acc[n], false, false);
    }

#pragma unroll
    for (int n = 0; n < 8; ++n) {
#pragma unroll
        for (int j = 0; j < 8; ++j) {
            const int grow = rbase + wave * 16 + j + half * 8;
            const int gcol = nbase + n * 16 + lm;
            const float v = acc[n][j] + bias[gcol];
            if (mode == 2) {
                outf[(size_t)grow * 1024 + gcol] = v;
            } else {
                const int b = grow >> 11, s = grow & 2047;
                const int h = gcol >> 6,  d = gcol & 63;
                const size_t idx = (mode == 0)
                    ? (((size_t)(b * NHEADS + h) * S_LEN + s) * DEPTH + d)
                    : (((size_t)(b * NHEADS + h) * DEPTH + d) * S_LEN + s);
                outb[idx] = f2bf_u(v);
            }
        }
    }
}

// Flash attention: 1 wave per (b, h, 16-query tile). Qh/Kh: [B,H,S,64], Vt: [B,H,64,S].
__global__ __launch_bounds__(32) void flash_attn(
    const unsigned short* __restrict__ Qh, const unsigned short* __restrict__ Kh,
    const unsigned short* __restrict__ Vt, const float* __restrict__ mask,
    unsigned short* __restrict__ ctx)
{
    __shared__ unsigned short Pl[16 * 48];   // P tile, row-major, stride 48

    const int lane = threadIdx.x;
    const int lm = lane & 15, half = lane >> 4;
    const int qt = blockIdx.x, h = blockIdx.y, b = blockIdx.z;
    const size_t bh = (size_t)(b * NHEADS + h);

    const unsigned short* Qbase = Qh + (bh * S_LEN + qt * 16) * DEPTH;
    const float* mbase = mask + (size_t)b * S_LEN;

    const int off = half * 8;     // A-fragment K offset
    const int k2  = half * 8;     // B-fragment K offset

    // per-lane fragment base pointers (per-iteration deltas become immediates)
    const unsigned short* kp = Kh + bh * S_LEN * DEPTH + lm * DEPTH + k2;
    const unsigned short* vp = Vt + bh * DEPTH * S_LEN + (size_t)lm * S_LEN + k2;

    Frag16 qf[2];
#pragma unroll
    for (int c = 0; c < 2; ++c) {
        const unsigned short* p = Qbase + lm * DEPTH + c * 32 + off;
        qf[c].q[0] = *(const u32x4*)p;
        qf[c].q[1] = *(const u32x4*)(p + 16);
    }

    v8f z = {0.f,0.f,0.f,0.f,0.f,0.f,0.f,0.f};
    v8f o[4];
    float m_i[8], l_i[8];
#pragma unroll
    for (int n = 0; n < 4; ++n) o[n] = z;
#pragma unroll
    for (int j = 0; j < 8; ++j) { m_i[j] = -1e30f; l_i[j] = 0.f; }

    for (int jb = 0; jb < S_LEN; jb += 32) {
        // batch-load 4 K fragments: [tile t][depth chunk c]
        const unsigned short* kj = kp + (size_t)jb * DEPTH;
        Frag16 kf[4];
#pragma unroll
        for (int t = 0; t < 2; ++t)
#pragma unroll
            for (int c = 0; c < 2; ++c) {
                const unsigned short* p = kj + t * (16 * DEPTH) + c * 32;
                kf[t * 2 + c].q[0] = *(const u32x4*)p;
                kf[t * 2 + c].q[1] = *(const u32x4*)(p + 16);
            }
        v8f s0 = z, s1 = z;
        s0 = __builtin_amdgcn_wmma_f32_16x16x32_bf16(false, qf[0].v, false, kf[0].v, (short)0, s0, false, false);
        s1 = __builtin_amdgcn_wmma_f32_16x16x32_bf16(false, qf[0].v, false, kf[2].v, (short)0, s1, false, false);
        s0 = __builtin_amdgcn_wmma_f32_16x16x32_bf16(false, qf[1].v, false, kf[1].v, (short)0, s0, false, false);
        s1 = __builtin_amdgcn_wmma_f32_16x16x32_bf16(false, qf[1].v, false, kf[3].v, (short)0, s1, false, false);

        const float mv0 = mbase[jb + lm]      * -1e10f;
        const float mv1 = mbase[jb + 16 + lm] * -1e10f;

        float p0a[8], p1a[8], alpha[8];
#pragma unroll
        for (int j = 0; j < 8; ++j) {
            float a = s0[j] * 0.125f + mv0;   // 1/sqrt(64)
            float c = s1[j] * 0.125f + mv1;
            float r = fmaxf(a, c);            // row max across 32 keys
            r = fmaxf(r, __shfl_xor(r, 1));
            r = fmaxf(r, __shfl_xor(r, 2));
            r = fmaxf(r, __shfl_xor(r, 4));
            r = fmaxf(r, __shfl_xor(r, 8));
            const float mn = fmaxf(m_i[j], r);
            alpha[j] = __expf(m_i[j] - mn);
            float e0 = __expf(a - mn), e1 = __expf(c - mn);
            p0a[j] = e0; p1a[j] = e1;
            float rs = e0 + e1;
            rs += __shfl_xor(rs, 1);
            rs += __shfl_xor(rs, 2);
            rs += __shfl_xor(rs, 4);
            rs += __shfl_xor(rs, 8);
            l_i[j] = l_i[j] * alpha[j] + rs;
            m_i[j] = mn;
        }

        // batch-load 4 V fragments while softmax math retires
        Frag16 vf[4];
#pragma unroll
        for (int n = 0; n < 4; ++n) {
            const unsigned short* p = vp + jb + n * (16 * S_LEN);
            vf[n].q[0] = *(const u32x4*)p;
            vf[n].q[1] = *(const u32x4*)(p + 16);
        }

#pragma unroll
        for (int n = 0; n < 4; ++n)
#pragma unroll
            for (int j = 0; j < 8; ++j) o[n][j] *= alpha[j];

        // C-layout -> A-layout transpose of P through LDS
#pragma unroll
        for (int j = 0; j < 8; ++j) {
            const int m = j + half * 8;
            Pl[m * 48 + lm]      = f2bf_u(p0a[j]);
            Pl[m * 48 + 16 + lm] = f2bf_u(p1a[j]);
        }
        __syncthreads();
        Frag16 pf;
        pf.q[0] = *(const u32x4*)&Pl[lm * 48 + off];
        pf.q[1] = *(const u32x4*)&Pl[lm * 48 + off + 16];
#pragma unroll
        for (int n = 0; n < 4; ++n)
            o[n] = __builtin_amdgcn_wmma_f32_16x16x32_bf16(
                false, pf.v, false, vf[n].v, (short)0, o[n], false, false);
        __syncthreads();
    }

    // normalize + store merged-head context [B,S,D] bf16
#pragma unroll
    for (int n = 0; n < 4; ++n)
#pragma unroll
        for (int j = 0; j < 8; ++j) {
            const int m = j + half * 8;
            const size_t idx = ((size_t)b * S_LEN + qt * 16 + m) * D_MODEL
                             + h * DEPTH + n * 16 + lm;
            ctx[idx] = f2bf_u(o[n][j] / l_i[j]);
        }
}

extern "C" void kernel_launch(void* const* d_in, const int* in_sizes, int n_in,
                              void* d_out, int out_size, void* d_ws, size_t ws_size,
                              hipStream_t stream) {
    const float* q    = (const float*)d_in[0];
    const float* k    = (const float*)d_in[1];
    const float* v    = (const float*)d_in[2];
    const float* mask = (const float*)d_in[3];
    const float* Wq   = (const float*)d_in[4];
    const float* bq   = (const float*)d_in[5];
    const float* Wk   = (const float*)d_in[6];
    const float* bk   = (const float*)d_in[7];
    const float* Wv   = (const float*)d_in[8];
    const float* bv   = (const float*)d_in[9];
    const float* Wo   = (const float*)d_in[10];
    const float* bo   = (const float*)d_in[11];

    const size_t SZA = (size_t)8192 * 1024 * 2;   // 16 MB bf16 activation
    const size_t SZW = (size_t)1024 * 1024 * 2;   // 2 MB bf16 weight
    char* ws = (char*)d_ws;
    unsigned short* qb  = (unsigned short*)(ws);
    unsigned short* kb  = (unsigned short*)(ws + 1 * SZA);
    unsigned short* vb  = (unsigned short*)(ws + 2 * SZA);
    unsigned short* wqt = (unsigned short*)(ws + 3 * SZA);
    unsigned short* wkt = (unsigned short*)(ws + 3 * SZA + 1 * SZW);
    unsigned short* wvt = (unsigned short*)(ws + 3 * SZA + 2 * SZW);
    unsigned short* wot = (unsigned short*)(ws + 3 * SZA + 3 * SZW);
    unsigned short* Qhp = (unsigned short*)(ws + 3 * SZA + 4 * SZW);
    unsigned short* Khp = (unsigned short*)(ws + 4 * SZA + 4 * SZW);
    unsigned short* Vtp = (unsigned short*)(ws + 5 * SZA + 4 * SZW);
    unsigned short* ctx = (unsigned short*)(ws + 6 * SZA + 4 * SZW);

    const int NA = 8192 * 1024, NW = 1024 * 1024;
    cvt_f32_bf16<<<(NA + 255) / 256, 256, 0, stream>>>(q, qb, NA);
    cvt_f32_bf16<<<(NA + 255) / 256, 256, 0, stream>>>(k, kb, NA);
    cvt_f32_bf16<<<(NA + 255) / 256, 256, 0, stream>>>(v, vb, NA);
    cvt_f32_bf16_T<<<NW / 256, 256, 0, stream>>>(Wq, wqt);
    cvt_f32_bf16_T<<<NW / 256, 256, 0, stream>>>(Wk, wkt);
    cvt_f32_bf16_T<<<NW / 256, 256, 0, stream>>>(Wv, wvt);
    cvt_f32_bf16_T<<<NW / 256, 256, 0, stream>>>(Wo, wot);

    dim3 gg(1024 / 128, 8192 / 128);  // (8, 64)
    wmma_gemm<<<gg, 256, 0, stream>>>(qb, wqt, bq, Qhp, nullptr, 0);
    wmma_gemm<<<gg, 256, 0, stream>>>(kb, wkt, bk, Khp, nullptr, 0);
    wmma_gemm<<<gg, 256, 0, stream>>>(vb, wvt, bv, Vtp, nullptr, 1);

    flash_attn<<<dim3(S_LEN / 16, NHEADS, 4), 32, 0, stream>>>(Qhp, Khp, Vtp, mask, ctx);

    wmma_gemm<<<gg, 256, 0, stream>>>(ctx, wot, bo, nullptr, (float*)d_out, 2);
}
